// IntensityTransformNet_34720515621480
// MI455X (gfx1250) — matrix operand order repaired and checked by reading, not compiled, not run on hardware
//
#include <hip/hip_runtime.h>
#include <hip/hip_bf16.h>

typedef __attribute__((ext_vector_type(16))) _Float16 v16h;
typedef __attribute__((ext_vector_type(8)))  float    v8f;
typedef __attribute__((ext_vector_type(4)))  float    f4;

__device__ __forceinline__ int imin(int a, int b) { return a < b ? a : b; }
__device__ __forceinline__ int imax(int a, int b) { return a > b ? a : b; }

// ---------------------------------------------------------------------------
// Generic WMMA row-tile GEMM helper (wave32, v_wmma_f32_16x16x32_f16).
// A fragment built per-lane from a functor getA(rowInTile, k) -> float.
// B fragment read from W in [N][K] row-major f16 (contiguous K run per lane).
// VGPR layouts used here (C/D as in CDNA5 ISA 7.12.2):
//   A 16x32: lane L -> row L&15; half h: K=(L>=16?16:0)+h (same order as B)
//   B 32x16: lane L -> col L&15; half h: K=(L>=16?16:0)+h
//   C/D:     VGPR r: lanes 0-15 -> M=r, lanes 16-31 -> M=r+8; N=lane&15
// ---------------------------------------------------------------------------
template <typename AF>
__device__ __forceinline__ v8f wmma_rowtile(AF&& getA,
                                            const _Float16* __restrict__ W,
                                            const int K, const int n0) {
  const int lane = (int)(threadIdx.x & 31u);
  const int rsel = lane >> 4;   // 0 or 1
  const int l15  = lane & 15;
  v8f acc = {};
  const _Float16* wrow = W + (size_t)(n0 + l15) * (size_t)K;
  for (int k0 = 0; k0 < K; k0 += 32) {
    v16h a;
    const int ka = k0 + rsel * 16;
#pragma unroll
    for (int h = 0; h < 8; ++h) {
      const int kA0 = ka + h;
      const int kA1 = ka + 8 + h;
      a[h]     = (_Float16)((kA0 < K) ? getA(l15, kA0) : 0.0f);
      a[h + 8] = (_Float16)((kA1 < K) ? getA(l15, kA1) : 0.0f);
    }
    v16h b;
    const int kb = k0 + rsel * 16;
#pragma unroll
    for (int h = 0; h < 16; ++h) {
      const int kk = kb + h;
      b[h] = (kk < K) ? wrow[kk] : (_Float16)0.0f;
    }
    acc = __builtin_amdgcn_wmma_f32_16x16x32_f16(
        /*neg_a=*/false, a, /*neg_b=*/false, b,
        /*c_mod=*/(short)0, acc, /*reuse_a=*/false, /*reuse_b=*/false);
  }
  return acc;
}

// ---------------------------------------------------------------------------
// float -> f16 weight conversion
// ---------------------------------------------------------------------------
__global__ void k_f2h(const float* __restrict__ s, _Float16* __restrict__ d, int n) {
  int i = blockIdx.x * blockDim.x + threadIdx.x;
  if (i < n) d[i] = (_Float16)s[i];
}

// ---------------------------------------------------------------------------
// Antialiased bilinear resize 2048x2048 -> 224x224 (jax.image.resize default:
// half-pixel mapping, triangle kernel of half-width S = 2048/224, boundary
// renormalization). Separable: H pass then V pass.
// ---------------------------------------------------------------------------
#define RS_S (2048.0f / 224.0f)

__global__ void k_resize_h(const float* __restrict__ x, float* __restrict__ tmp) {
  int i = blockIdx.x * blockDim.x + threadIdx.x;       // [12][2048][224]
  if (i >= 12 * 2048 * 224) return;
  int ox = i % 224;  int t = i / 224;
  int y = t % 2048;  int plane = t / 2048;
  float center = (ox + 0.5f) * RS_S - 0.5f;
  int lo = imax((int)ceilf(center - RS_S), 0);
  int hi = imin((int)floorf(center + RS_S), 2047);
  const float* row = x + ((size_t)plane * 2048 + (size_t)y) * 2048;
  float acc = 0.f, wsum = 0.f;
  for (int ix = lo; ix <= hi; ++ix) {
    float w = 1.0f - fabsf((float)ix - center) * (1.0f / RS_S);
    w = fmaxf(w, 0.0f);
    acc += w * row[ix];
    wsum += w;
  }
  tmp[i] = acc / wsum;
}

__global__ void k_resize_v(const float* __restrict__ tmp, float* __restrict__ low) {
  int i = blockIdx.x * blockDim.x + threadIdx.x;       // [12][224][224]
  if (i >= 12 * 224 * 224) return;
  int ox = i % 224; int t = i / 224;
  int oy = t % 224; int plane = t / 224;
  float center = (oy + 0.5f) * RS_S - 0.5f;
  int lo = imax((int)ceilf(center - RS_S), 0);
  int hi = imin((int)floorf(center + RS_S), 2047);
  float acc = 0.f, wsum = 0.f;
  for (int iy = lo; iy <= hi; ++iy) {
    float w = 1.0f - fabsf((float)iy - center) * (1.0f / RS_S);
    w = fmaxf(w, 0.0f);
    acc += w * tmp[((size_t)plane * 2048 + (size_t)iy) * 224 + ox];
    wsum += w;
  }
  low[i] = acc / wsum;                                 // [b][c][oy][ox]
}

// ---------------------------------------------------------------------------
// conv1: 3->32ch, 7x7, stride 3, replicate pad 1, ReLU. Implicit GEMM:
// M = 4*74*74 = 21904 (1369 tiles of 16), K = 147, N = 32 (2 waves/WG).
// ---------------------------------------------------------------------------
__global__ void k_conv1(const float* __restrict__ low, const _Float16* __restrict__ w1h,
                        const float* __restrict__ b1, float* __restrict__ f1) {
  const int m0 = blockIdx.x * 16;
  const int n0 = (threadIdx.x >> 5) * 16;              // wave -> N tile
  auto getA = [&](int mr, int k) -> float {
    int m = m0 + mr;
    int b = m / 5476, r = m % 5476;
    int oy = r / 74, ox = r % 74;
    int c = k / 49, rk = k % 49;
    int ky = rk / 7, kx = rk % 7;
    int iy = imin(imax(oy * 3 - 1 + ky, 0), 223);
    int ix = imin(imax(ox * 3 - 1 + kx, 0), 223);
    return low[(((size_t)b * 3 + c) * 224 + iy) * 224 + ix];
  };
  v8f acc = wmma_rowtile(getA, w1h, 147, n0);
  const int lane = (int)(threadIdx.x & 31u);
  const int rsel = lane >> 4;
  const int ncol = n0 + (lane & 15);
  const float bias = b1[ncol];
#pragma unroll
  for (int r = 0; r < 8; ++r) {
    int m = m0 + r + rsel * 8;
    int b = m / 5476, rr = m % 5476;
    int oy = rr / 74, ox = rr % 74;
    f1[(((size_t)b * 32 + ncol) * 74 + oy) * 74 + ox] = fmaxf(acc[r] + bias, 0.0f);
  }
}

// ---------------------------------------------------------------------------
// conv2: 32->64ch, 5x5, stride 2, replicate pad 1, ReLU. Implicit GEMM:
// M = 4*36*36 = 5184 (324 tiles of 16), K = 800, N = 64 (4 waves/WG).
// ---------------------------------------------------------------------------
__global__ void k_conv2(const float* __restrict__ f1, const _Float16* __restrict__ w2h,
                        const float* __restrict__ b2, float* __restrict__ f2) {
  const int m0 = blockIdx.x * 16;
  const int n0 = (threadIdx.x >> 5) * 16;
  auto getA = [&](int mr, int k) -> float {
    int m = m0 + mr;
    int b = m / 1296, r = m % 1296;
    int oy = r / 36, ox = r % 36;
    int c = k / 25, rk = k % 25;
    int ky = rk / 5, kx = rk % 5;
    int iy = imin(imax(oy * 2 - 1 + ky, 0), 73);
    int ix = imin(imax(ox * 2 - 1 + kx, 0), 73);
    return f1[(((size_t)b * 32 + c) * 74 + iy) * 74 + ix];
  };
  v8f acc = wmma_rowtile(getA, w2h, 800, n0);
  const int lane = (int)(threadIdx.x & 31u);
  const int rsel = lane >> 4;
  const int ncol = n0 + (lane & 15);
  const float bias = b2[ncol];
#pragma unroll
  for (int r = 0; r < 8; ++r) {
    int m = m0 + r + rsel * 8;
    int b = m / 1296, rr = m % 1296;
    int oy = rr / 36, ox = rr % 36;
    f2[(((size_t)b * 64 + ncol) * 36 + oy) * 36 + ox] = fmaxf(acc[r] + bias, 0.0f);
  }
}

// ---------------------------------------------------------------------------
// 9x9/9 max pool + flatten: [4,64,36,36] -> feat [4,1024]
// ---------------------------------------------------------------------------
__global__ void k_pool(const float* __restrict__ f2, float* __restrict__ feat) {
  int i = blockIdx.x * blockDim.x + threadIdx.x;       // 4096
  if (i >= 4096) return;
  int b = i >> 10;
  int fidx = i & 1023;                                 // c*16 + py*4 + px
  int c = fidx >> 4;
  int py = (fidx >> 2) & 3, px = fidx & 3;
  const float* pl = f2 + ((size_t)b * 64 + c) * 36 * 36;
  float mx = -3.402823466e38f;
  for (int dy = 0; dy < 9; ++dy)
    for (int dx = 0; dx < 9; ++dx)
      mx = fmaxf(mx, pl[(py * 9 + dy) * 36 + px * 9 + dx]);
  feat[(size_t)b * 1024 + fidx] = mx;
}

// ---------------------------------------------------------------------------
// FC: curves[4,768] = feat[4,1024] @ fc_w[768,1024]^T + fc_b.
// One wave per 16x16 tile, M padded 4->16, grid.x = 768/16 = 48.
// ---------------------------------------------------------------------------
__global__ void k_fc(const float* __restrict__ feat, const _Float16* __restrict__ fwh,
                     const float* __restrict__ fcb, float* __restrict__ curves) {
  const int n0 = blockIdx.x * 16;
  auto getA = [&](int mr, int k) -> float {
    return (mr < 4) ? feat[(size_t)mr * 1024 + k] : 0.0f;
  };
  v8f acc = wmma_rowtile(getA, fwh, 1024, n0);
  const int lane = (int)(threadIdx.x & 31u);
  const int rsel = lane >> 4;
  const int ncol = n0 + (lane & 15);
  const float bias = fcb[ncol];
#pragma unroll
  for (int r = 0; r < 8; ++r) {
    int m = r + rsel * 8;
    if (m < 4) curves[(size_t)m * 768 + ncol] = acc[r] + bias;
  }
}

// ---------------------------------------------------------------------------
// Full-res LUT apply (dominant, HBM-bound pass). Curve packed as
// (c[i], c[min(i+1,255)]) float2 pairs in LDS -> one ds_load_b64 per pixel.
// Non-temporal 128-bit streaming of x and out (both > L2, touched once).
// ---------------------------------------------------------------------------
__device__ __forceinline__ float lut_apply(const float2* lut, float xv) {
  float v = fminf(fmaxf(xv * 255.0f, 0.0f), 255.0f);
  float d = floorf(v);
  int di = (int)d;
  float w = v - d;
  float2 p = lut[di];
  return (p.x + (p.y - p.x) * w) * (1.0f / 255.0f);
}

__global__ void k_lut(const float* __restrict__ x, const float* __restrict__ curves,
                      float* __restrict__ out) {
  __shared__ float2 lut[256];
  const int plane = blockIdx.y;                        // b*3 + c, 0..11
  const float* cv = curves + (size_t)(plane / 3) * 768 + (size_t)(plane % 3) * 256;
  const int tid = threadIdx.x;                         // blockDim.x == 256
  lut[tid] = make_float2(cv[tid], cv[imin(tid + 1, 255)]);
  __syncthreads();

  const f4* __restrict__ xin = (const f4*)(x + (size_t)plane * 4194304);
  f4* __restrict__ op = (f4*)(out + (size_t)plane * 4194304);
  const int total = 1048576;                           // float4s per plane
  const int stride = gridDim.x * blockDim.x;
  for (int idx = blockIdx.x * blockDim.x + tid; idx < total; idx += stride) {
    f4 v = __builtin_nontemporal_load(xin + idx);
    f4 r;
#pragma unroll
    for (int c = 0; c < 4; ++c) r[c] = lut_apply(lut, v[c]);
    __builtin_nontemporal_store(r, op + idx);
  }
}

// ---------------------------------------------------------------------------
// Launch
// ---------------------------------------------------------------------------
extern "C" void kernel_launch(void* const* d_in, const int* in_sizes, int n_in,
                              void* d_out, int out_size, void* d_ws, size_t ws_size,
                              hipStream_t stream) {
  const float* x  = (const float*)d_in[0];  // [4,3,2048,2048]
  const float* w1 = (const float*)d_in[1];  // [32,3,7,7]
  const float* b1 = (const float*)d_in[2];  // [32]
  const float* w2 = (const float*)d_in[3];  // [64,32,5,5]
  const float* b2 = (const float*)d_in[4];  // [64]
  const float* fw = (const float*)d_in[5];  // [768,1024]
  const float* fb = (const float*)d_in[6];  // [768]
  float* out = (float*)d_out;

  char* p = (char*)d_ws;
  auto alloc = [&](size_t bytes) { char* r = p; p += (bytes + 255) & ~(size_t)255; return (void*)r; };
  float*    low    = (float*)   alloc(602112 * 4);  // [4,3,224,224]
  float*    f1     = (float*)   alloc(700928 * 4);  // [4,32,74,74]
  float*    f2     = (float*)   alloc(331776 * 4);  // [4,64,36,36]
  float*    feat   = (float*)   alloc(4096   * 4);  // [4,1024]
  float*    curves = (float*)   alloc(3072   * 4);  // [4,768]
  _Float16* w1h    = (_Float16*)alloc(1568   * 2);
  _Float16* w2h    = (_Float16*)alloc(51200  * 2);
  _Float16* fwh    = (_Float16*)alloc(786432 * 2);

  // H-pass temp (22 MB) lives in d_out: dead before k_lut overwrites all of out.
  float* tmp = (float*)d_out;                         // [12,2048,224]

  k_f2h<<<(1568   + 255) / 256, 256, 0, stream>>>(w1, w1h, 1568);
  k_f2h<<<(51200  + 255) / 256, 256, 0, stream>>>(w2, w2h, 51200);
  k_f2h<<<(786432 + 255) / 256, 256, 0, stream>>>(fw, fwh, 786432);

  k_resize_h<<<(12 * 2048 * 224 + 255) / 256, 256, 0, stream>>>(x, tmp);
  k_resize_v<<<(12 * 224 * 224 + 255) / 256, 256, 0, stream>>>(tmp, low);

  k_conv1<<<1369, 64,  0, stream>>>(low, w1h, b1, f1);
  k_conv2<<<324,  128, 0, stream>>>(f1, w2h, b2, f2);
  k_pool <<<16,   256, 0, stream>>>(f2, feat);
  k_fc   <<<48,   32,  0, stream>>>(feat, fwh, fb, curves);

  k_lut<<<dim3(512, 12), 256, 0, stream>>>(x, curves, out);
}
